// AttentionBlock_76295799046627
// MI455X (gfx1250) — compile-verified, matrix operand order checked by reading, not run on hardware
//
#include <hip/hip_runtime.h>

typedef __attribute__((ext_vector_type(16))) _Float16 v16h;
typedef __attribute__((ext_vector_type(8)))  _Float16 v8h;
typedef __attribute__((ext_vector_type(8)))  float    v8f;

#define HW   4096
#define CCH  256
#define NPC  (HW * CCH)   // per-batch elements of [hw, c] tensors
#define QW   4            // query-tile waves per flash block

// ---------------------------------------------------------------------------
// Load a 16x32 f16 WMMA A/B fragment (NT): `p` points at the start of this
// lane's row (K-contiguous). Per ISA 7.12.2, lane (l<16) holds K = {0..7,
// 16..23}, lane (l>=16) holds K = {8..15, 24..31}. Works for global or LDS.
// ---------------------------------------------------------------------------
__device__ __forceinline__ v16h load_frag(const _Float16* p, int half) {
  const _Float16* q = p + half * 8;
  v8h lo = *(const v8h*)(q);        // K = half*8 .. half*8+7
  v8h hi = *(const v8h*)(q + 16);   // K = 16 + half*8 ..
  return __builtin_shufflevector(lo, hi, 0,1,2,3,4,5,6,7,8,9,10,11,12,13,14,15);
}

__device__ __forceinline__ float redmax16(float v) {
#pragma unroll
  for (int m = 8; m >= 1; m >>= 1) v = fmaxf(v, __shfl_xor(v, m, 32));
  return v;
}
__device__ __forceinline__ float redsum16(float v) {
#pragma unroll
  for (int m = 8; m >= 1; m >>= 1) v += __shfl_xor(v, m, 32);
  return v;
}

// Async global -> LDS copy of 16 bytes per lane (CDNA5 ASYNCcnt path).
__device__ __forceinline__ void async_b128(const _Float16* gsrc, void* lds_dst) {
  unsigned dst = (unsigned)(uintptr_t)lds_dst;  // low 32 bits of flat = LDS offset
  asm volatile("global_load_async_to_lds_b128 %0, %1, off"
               :: "v"(dst), "v"(gsrc) : "memory");
}
__device__ __forceinline__ void wait_async0() {
  asm volatile("s_wait_asynccnt 0" ::: "memory");
}

// ---------------------------------------------------------------------------
// f32 -> f16 conversion (weights)
// ---------------------------------------------------------------------------
__global__ void f32_to_f16_kernel(const float* __restrict__ in,
                                  _Float16* __restrict__ out, int n) {
  int i = blockIdx.x * blockDim.x + threadIdx.x;
  if (i < n) out[i] = (_Float16)in[i];
}

// ---------------------------------------------------------------------------
// GroupNorm over x[b, c, hw] (8 groups of 32 channels) + transposed f16 store
// xn16[b, hw, c]. One workgroup (256 thr) per (batch, group).
// ---------------------------------------------------------------------------
__global__ __launch_bounds__(256)
void groupnorm_kernel(const float* __restrict__ x,
                      const float* __restrict__ gamma,
                      const float* __restrict__ beta,
                      _Float16* __restrict__ xn16) {
  int bg = blockIdx.x;            // 0..31
  int b  = bg >> 3, g = bg & 7;
  int c0 = g * 32;
  const float* xb = x + ((long)b * CCH + c0) * HW;
  int tid = threadIdx.x;

  float s = 0.f, ss = 0.f;
  for (int i = tid; i < 32 * HW; i += 256) {
    float v = xb[i];
    s += v; ss += v * v;
  }
  __shared__ float red[256], red2[256];
  red[tid] = s; red2[tid] = ss;
  __syncthreads();
  for (int off = 128; off > 0; off >>= 1) {
    if (tid < off) { red[tid] += red[tid + off]; red2[tid] += red2[tid + off]; }
    __syncthreads();
  }
  float mean = red[0] * (1.0f / 131072.0f);
  float var  = red2[0] * (1.0f / 131072.0f) - mean * mean;
  float rstd = rsqrtf(var + 1e-5f);

  __shared__ float tile[32][65];
  _Float16* outb = xn16 + (long)b * NPC;
  float gam = gamma[c0 + (tid & 31)];
  float bet = beta [c0 + (tid & 31)];
  for (int p0 = 0; p0 < HW; p0 += 64) {
#pragma unroll
    for (int it = 0; it < 8; ++it) {      // coalesced read along p
      int cl = (tid >> 6) + it * 4;
      int pl = tid & 63;
      tile[cl][pl] = xb[(long)cl * HW + p0 + pl];
    }
    __syncthreads();
#pragma unroll
    for (int it = 0; it < 8; ++it) {      // coalesced write along c
      int pl = (tid >> 5) + it * 8;
      int cl = tid & 31;
      float v = (tile[cl][pl] - mean) * rstd * gam + bet;
      outb[(long)(p0 + pl) * CCH + c0 + cl] = (_Float16)v;
    }
    __syncthreads();
  }
}

// ---------------------------------------------------------------------------
// NT GEMM: C[m,n] = sum_k A[m,k] * B[n,k] + bias, K = 256, lda = ldb = 256.
// Block = 8 waves; wave w owns rows [by*128 + w*16, +16), cols [bx*64, +64).
// Software-pipelined: fragments for k0+32 are loaded before the WMMAs of k0.
// ---------------------------------------------------------------------------
template <bool BIAS_ROW, bool OUT_RESID>
__global__ __launch_bounds__(256)
void gemm_nt_kernel(const _Float16* __restrict__ A, long sA,
                    const _Float16* __restrict__ B, long sB,
                    const float* __restrict__ bias,
                    void* __restrict__ Cout, long sC, int ldc,
                    const float* __restrict__ resid, int M) {
  int bz = blockIdx.z;
  const _Float16* Ab = A + (long)bz * sA;
  const _Float16* Bb = B + (long)bz * sB;

  int lane = threadIdx.x & 31;
  int wave = threadIdx.x >> 5;
  int half = lane >> 4, r = lane & 15;
  int mtile = blockIdx.y * 128 + wave * 16;
  int n0    = blockIdx.x * 64;
  if (mtile >= M) return;                        // wave-uniform

  const _Float16* arow = Ab + (long)(mtile + r) * 256;
  const _Float16* brow0 = Bb + (long)(n0 + r) * 256;

  v8f acc[4] = {};
  v16h af = load_frag(arow, half);
  v16h bf[4];
#pragma unroll
  for (int j = 0; j < 4; ++j) bf[j] = load_frag(brow0 + (long)j * 16 * 256, half);

#pragma unroll
  for (int k0 = 0; k0 < 256; k0 += 32) {
    v16h afn = af;
    v16h bfn[4] = {bf[0], bf[1], bf[2], bf[3]};
    if (k0 + 32 < 256) {                         // prefetch next k-chunk
      afn = load_frag(arow + k0 + 32, half);
#pragma unroll
      for (int j = 0; j < 4; ++j)
        bfn[j] = load_frag(brow0 + (long)j * 16 * 256 + k0 + 32, half);
    }
#pragma unroll
    for (int j = 0; j < 4; ++j)
      acc[j] = __builtin_amdgcn_wmma_f32_16x16x32_f16(
          false, af, false, bf[j], (short)0, acc[j], false, false);
    af = afn;
#pragma unroll
    for (int j = 0; j < 4; ++j) bf[j] = bfn[j];
  }

#pragma unroll
  for (int j = 0; j < 4; ++j) {
#pragma unroll
    for (int i = 0; i < 8; ++i) {
      int row = mtile + i + 8 * half;            // C layout: M = i + 8*half
      int col = n0 + j * 16 + r;                 //           N = lane & 15
      float v = acc[j][i] + (BIAS_ROW ? bias[row] : bias[col]);
      if (OUT_RESID) {
        float* o = (float*)Cout + (long)bz * sC;
        o[(long)row * ldc + col] = v + resid[(long)bz * sC + (long)row * ldc + col];
      } else {
        _Float16* o = (_Float16*)Cout + (long)bz * sC;
        o[(long)row * ldc + col] = (_Float16)v;
      }
    }
  }
}

// ---------------------------------------------------------------------------
// Flash attention. Block = 4 waves, each owning a 16-query tile; K/V 32-key
// chunks are double-buffered in LDS via global_load_async_to_lds_b128 and
// shared by all 4 waves. Q lives in VGPRs. Fragment loads are software-
// pipelined one WMMA-pair ahead so s_wait_dscnt never covers a fresh load.
// ---------------------------------------------------------------------------
#define KPAD 264   // 256 + 8 halves: row stride 528 B -> base banks 4*r mod 64
#define VPAD 40    // 32 + 8 halves:  row stride 80 B  -> conflict-free b128

__global__ __launch_bounds__(32 * QW)
void flash_attn_kernel(const _Float16* __restrict__ q16,
                       const _Float16* __restrict__ k16,
                       const _Float16* __restrict__ vt16,
                       _Float16* __restrict__ o16) {
  int b = blockIdx.y;
  long bofs = (long)b * NPC;
  const _Float16* Q  = q16  + bofs;
  const _Float16* Kp = k16  + bofs;
  const _Float16* Vt = vt16 + bofs;    // [256][4096]

  int tid  = threadIdx.x;
  int wave = tid >> 5;
  int lane = tid & 31;
  int half = lane >> 4, r = lane & 15;
  int qt   = blockIdx.x * QW + wave;   // 16-row query tile
  const float scale = 0.0625f;         // 1/sqrt(256)

  __shared__ _Float16 kbuf[2][32][KPAD];   // [buf][key][feat]
  __shared__ _Float16 vbuf[2][256][VPAD];  // [buf][chan][key]
  __shared__ _Float16 Pt[QW][16 * 32];

  // cooperative async fill of one 32-key K/V chunk (all 128 threads)
  auto fill = [&](int buf, int kc) {
#pragma unroll
    for (int u0 = 0; u0 < 8; ++u0) {         // K tile: 32 rows x 32 16B-segs
      int u = tid + u0 * 128;
      int row = u >> 5, seg = u & 31;
      async_b128(Kp + (long)(kc + row) * 256 + seg * 8, &kbuf[buf][row][seg * 8]);
    }
#pragma unroll
    for (int u0 = 0; u0 < 8; ++u0) {         // V tile: 256 rows x 4 16B-segs
      int u = tid + u0 * 128;
      int ch = u >> 2, seg = u & 3;
      async_b128(Vt + (long)ch * HW + kc + seg * 8, &vbuf[buf][ch][seg * 8]);
    }
  };

  // Q fragments resident in VGPRs (8 x v16h = 64 VGPRs)
  const _Float16* qrow = Q + (long)(qt * 16 + r) * 256;
  v16h q[8];
#pragma unroll
  for (int kf = 0; kf < 8; ++kf) q[kf] = load_frag(qrow + kf * 32, half);

  v8f o[16];
#pragma unroll
  for (int nt = 0; nt < 16; ++nt) o[nt] = (v8f){};
  float mrow[8], lrow[8];
#pragma unroll
  for (int j = 0; j < 8; ++j) { mrow[j] = -1e30f; lrow[j] = 0.f; }

  fill(0, 0);
  wait_async0();
  __syncthreads();
  int cur = 0;

#pragma unroll 1
  for (int kc = 0; kc < HW; kc += 32) {
    if (kc + 32 < HW) fill(cur ^ 1, kc + 32);  // DMA next chunk during compute

    // ---- S = Q * K^T for keys [kc, kc+32), operands from LDS, pipelined ----
    v8f s0 = (v8f){}, s1 = (v8f){};
    const _Float16* kb0 = &kbuf[cur][r][0];
    const _Float16* kb1 = &kbuf[cur][16 + r][0];
    v16h b0 = load_frag(kb0, half);
    v16h b1 = load_frag(kb1, half);
#pragma unroll
    for (int kf = 0; kf < 8; ++kf) {
      v16h nb0 = b0, nb1 = b1;
      if (kf < 7) {                              // prefetch kf+1 before WMMAs
        nb0 = load_frag(kb0 + (kf + 1) * 32, half);
        nb1 = load_frag(kb1 + (kf + 1) * 32, half);
      }
      s0 = __builtin_amdgcn_wmma_f32_16x16x32_f16(false, q[kf], false, b0,
                                                  (short)0, s0, false, false);
      s1 = __builtin_amdgcn_wmma_f32_16x16x32_f16(false, q[kf], false, b1,
                                                  (short)0, s1, false, false);
      b0 = nb0; b1 = nb1;
    }
    // ---- online softmax (row = j + 8*half, col = r in 16-lane group) ----
#pragma unroll
    for (int j = 0; j < 8; ++j) {
      float t0 = s0[j] * scale, t1 = s1[j] * scale;
      float mx = redmax16(fmaxf(t0, t1));
      float mnew = fmaxf(mrow[j], mx);
      float corr = __expf(mrow[j] - mnew);
      float p0 = __expf(t0 - mnew);
      float p1 = __expf(t1 - mnew);
      lrow[j] = lrow[j] * corr + redsum16(p0 + p1);
      mrow[j] = mnew;
#pragma unroll
      for (int nt = 0; nt < 16; ++nt) o[nt][j] *= corr;
      Pt[wave][(j + 8 * half) * 32 + r]      = (_Float16)p0;
      Pt[wave][(j + 8 * half) * 32 + 16 + r] = (_Float16)p1;
    }
    // ---- O += P * V (P via per-wave LDS round-trip; V from LDS), pipelined --
    v16h pa = load_frag(&Pt[wave][r * 32], half);
    v16h vb = load_frag(&vbuf[cur][r][0], half);
#pragma unroll
    for (int nt = 0; nt < 16; ++nt) {
      v16h nvb = vb;
      if (nt < 15) nvb = load_frag(&vbuf[cur][(nt + 1) * 16 + r][0], half);
      o[nt] = __builtin_amdgcn_wmma_f32_16x16x32_f16(false, pa, false, vb,
                                                     (short)0, o[nt], false, false);
      vb = nvb;
    }
    wait_async0();     // next buffer's DMA complete (own share)
    __syncthreads();   // everyone's DMA done & buf[cur] reads retired
    cur ^= 1;
  }

  _Float16* ob = o16 + bofs;
#pragma unroll
  for (int j = 0; j < 8; ++j) {
    float inv = 1.0f / lrow[j];
#pragma unroll
    for (int nt = 0; nt < 16; ++nt) {
      ob[(long)(qt * 16 + j + 8 * half) * 256 + nt * 16 + r] =
          (_Float16)(o[nt][j] * inv);
    }
  }
}

// ---------------------------------------------------------------------------
extern "C" void kernel_launch(void* const* d_in, const int* in_sizes, int n_in,
                              void* d_out, int out_size, void* d_ws, size_t ws_size,
                              hipStream_t stream) {
  const float* x     = (const float*)d_in[0];
  const float* gamma = (const float*)d_in[1];
  const float* beta  = (const float*)d_in[2];
  const float* wq    = (const float*)d_in[3];
  const float* bq    = (const float*)d_in[4];
  const float* wk    = (const float*)d_in[5];
  const float* bk    = (const float*)d_in[6];
  const float* wv    = (const float*)d_in[7];
  const float* bv    = (const float*)d_in[8];
  const float* wp    = (const float*)d_in[9];
  const float* bp    = (const float*)d_in[10];

  _Float16* xn16 = (_Float16*)d_ws;          // [4][hw][c]
  _Float16* q16  = xn16 + 4L * NPC;          // [4][hw][c]
  _Float16* k16  = q16  + 4L * NPC;          // [4][hw][c]
  _Float16* vt16 = k16  + 4L * NPC;          // [4][c][hw]
  _Float16* o16  = vt16 + 4L * NPC;          // [4][hw][c]
  _Float16* wq16 = o16  + 4L * NPC;          // [256][256] each
  _Float16* wk16 = wq16 + 65536;
  _Float16* wv16 = wk16 + 65536;
  _Float16* wp16 = wv16 + 65536;

  f32_to_f16_kernel<<<256, 256, 0, stream>>>(wq, wq16, 65536);
  f32_to_f16_kernel<<<256, 256, 0, stream>>>(wk, wk16, 65536);
  f32_to_f16_kernel<<<256, 256, 0, stream>>>(wv, wv16, 65536);
  f32_to_f16_kernel<<<256, 256, 0, stream>>>(wp, wp16, 65536);

  groupnorm_kernel<<<32, 256, 0, stream>>>(x, gamma, beta, xn16);

  gemm_nt_kernel<false, false><<<dim3(4, 32, 4), 256, 0, stream>>>(
      xn16, NPC, wq16, 0, bq, q16, NPC, 256, nullptr, HW);
  gemm_nt_kernel<false, false><<<dim3(4, 32, 4), 256, 0, stream>>>(
      xn16, NPC, wk16, 0, bk, k16, NPC, 256, nullptr, HW);
  gemm_nt_kernel<true, false><<<dim3(64, 2, 4), 256, 0, stream>>>(
      wv16, 0, xn16, NPC, bv, vt16, NPC, HW, nullptr, CCH);

  flash_attn_kernel<<<dim3(HW / (16 * QW), 4), 32 * QW, 0, stream>>>(
      q16, k16, vt16, o16);

  gemm_nt_kernel<true, true><<<dim3(64, 2, 4), 256, 0, stream>>>(
      wp16, 0, o16, NPC, bp, d_out, NPC, HW, x, CCH);
}
